// RWKV_Tmix_x070_1889785610320
// MI455X (gfx1250) — matrix operand order repaired and checked by reading, not compiled, not run on hardware
//
#include <hip/hip_runtime.h>
#include <hip/hip_bf16.h>
#include <math.h>

// RWKV-7 Tmix forward for MI455X (gfx1250, wave32, WMMA).
// B=2, T=1024, C=2048, H=32, N=64.

#define C_DIM  2048
#define HEADS  32
#define HN     64
#define B_DIM  2
#define T_DIM  1024
#define M_ROWS (B_DIM * T_DIM)   // 2048 rows for all GEMMs (always tile-exact)
#define GN_EPS 6.4e-4f           // 1e-5 * 8^2

typedef __attribute__((ext_vector_type(16))) __bf16 v16bf;
typedef __attribute__((ext_vector_type(8)))  __bf16 v8bf;
typedef __attribute__((ext_vector_type(4)))  __bf16 v4bf;
typedef __attribute__((ext_vector_type(8)))  float  v8f;
typedef __attribute__((ext_vector_type(4)))  float  v4f;

__device__ __forceinline__ __bf16 f2bf(float f) {
  union { float f; unsigned u; } x; x.f = f;
  unsigned r = x.u + 0x7FFFu + ((x.u >> 16) & 1u);   // round-to-nearest-even
  unsigned short hv = (unsigned short)(r >> 16);
  __bf16 o; __builtin_memcpy(&o, &hv, 2);
  return o;
}
__device__ __forceinline__ v4bf f2bf4(v4f f) {
  v4bf o;
  o[0] = f2bf(f[0]); o[1] = f2bf(f[1]); o[2] = f2bf(f[2]); o[3] = f2bf(f[3]);
  return o;
}
__device__ __forceinline__ float sigmoidf_(float x) { return 1.0f / (1.0f + __expf(-x)); }

// ---------------------------------------------------------------------------
// GEMM  Y[M,N] = X[M,K] * W[K,N]  (fp32 in/out, bf16 WMMA compute, fp32 acc)
// Block 256 = 8 waves; tile 64x64, K-step 32; wave grid 4(m) x 2(n), each
// wave owns a 16x32 strip = 2 WMMA 16x16x32 tiles sharing the A fragment.
// B staged TRANSPOSED in LDS -> all fragments are contiguous ds_load_b128.
// Double-buffered LDS software pipeline: global loads for tile k+1 overlap
// the WMMAs of tile k; one barrier per K-step.
// Requires: M % 64 == 0, K % 32 == 0. GUARD_N handles N % 64 != 0 (N=32).
// ---------------------------------------------------------------------------
template <bool GUARD_N>
__global__ __launch_bounds__(256)
void gemm_bf16_wmma(const float* __restrict__ X, const float* __restrict__ W,
                    float* __restrict__ Y, int M, int N, int K)
{
  __builtin_assume(threadIdx.x < 256);
  __shared__ __align__(16) __bf16 As [2][64][40];   // row stride 80B (16B mult)
  __shared__ __align__(16) __bf16 BsT[2][64][40];   // [buf][n-col][k]

  const int tid  = threadIdx.x;
  const int lane = tid & 31;
  const int wave = tid >> 5;
  const int wm   = wave & 3;      // 0..3
  const int wn   = wave >> 2;     // 0..1
  const int mBlk = blockIdx.y * 64;
  const int nBlk = blockIdx.x * 64;
  const int m0   = wm * 16;
  const int n0   = wn * 32;

  // fixed per-thread staging coordinates (2 float4 for A, 2x4 strided for B)
  const int ar0 = tid >> 3;             // 0..31
  const int ar1 = ar0 + 32;             // 32..63
  const int ac  = (tid & 7) << 2;       // 0..28
  const int bc  = tid & 63;             // n-col in tile
  const int br0 = (tid >> 6) << 2;      // 0,4,8,12
  const int br1 = br0 + 16;             // 16,20,24,28
  const int gn  = nBlk + bc;
  const int gnc = GUARD_N ? (gn < N ? gn : 0) : gn;
  const float bmask = (!GUARD_N || gn < N) ? 1.0f : 0.0f;

  v8f acc0 = {0.f,0.f,0.f,0.f,0.f,0.f,0.f,0.f};
  v8f acc1 = acc0;

  v4f ra0, ra1, rb0, rb1;   // staging registers (tile in flight)

  auto loadG = [&](int k0) {
    ra0 = *(const v4f*)(X + (size_t)(mBlk + ar0) * K + (k0 + ac));
    ra1 = *(const v4f*)(X + (size_t)(mBlk + ar1) * K + (k0 + ac));
    #pragma unroll
    for (int j = 0; j < 4; ++j) rb0[j] = W[(size_t)(k0 + br0 + j) * N + gnc];
    #pragma unroll
    for (int j = 0; j < 4; ++j) rb1[j] = W[(size_t)(k0 + br1 + j) * N + gnc];
    if (GUARD_N) { rb0 *= bmask; rb1 *= bmask; }
  };
  auto storeL = [&](int buf) {
    *(v4bf*)&As [buf][ar0][ac]  = f2bf4(ra0);
    *(v4bf*)&As [buf][ar1][ac]  = f2bf4(ra1);
    *(v4bf*)&BsT[buf][bc][br0]  = f2bf4(rb0);
    *(v4bf*)&BsT[buf][bc][br1]  = f2bf4(rb1);
  };

  loadG(0);
  storeL(0);
  __syncthreads();

  const int nk = K >> 5;
  for (int kt = 0; kt < nk; ++kt) {
    const int cur = kt & 1;
    if (kt + 1 < nk) {
      loadG((kt + 1) << 5);                       // overlap with WMMA below
      if (kt + 2 < nk) {                          // L2 hint two tiles out
        __builtin_prefetch(&X[(size_t)(mBlk + ar0) * K + ((kt + 2) << 5) + ac], 0, 1);
        __builtin_prefetch(&W[(size_t)(((kt + 2) << 5) + br0) * N + gnc], 0, 1);
      }
    }

    // fragments: two ds_load_b128 each (ISA wave32 VGPR layouts)
    union F16 { v16bf v; v8bf h[2]; };
    F16 af, b0f, b1f;
    {
      const int row = m0 + (lane & 15);
      const int kb  = (lane >> 4) << 3;           // 0 | 8
      af.h[0] = *(const v8bf*)&As[cur][row][kb];
      af.h[1] = *(const v8bf*)&As[cur][row][16 + kb];
    }
    {
      const int col = n0 + (lane & 15);
      const int kb  = (lane >> 4) << 3;           // 0 | 8
      b0f.h[0] = *(const v8bf*)&BsT[cur][col][kb];
      b0f.h[1] = *(const v8bf*)&BsT[cur][col][16 + kb];
      b1f.h[0] = *(const v8bf*)&BsT[cur][col + 16][kb];
      b1f.h[1] = *(const v8bf*)&BsT[cur][col + 16][16 + kb];
    }
    acc0 = __builtin_amdgcn_wmma_f32_16x16x32_bf16(false, af.v, false, b0f.v,
                                                   (short)0, acc0, false, false);
    acc1 = __builtin_amdgcn_wmma_f32_16x16x32_bf16(false, af.v, false, b1f.v,
                                                   (short)0, acc1, false, false);

    if (kt + 1 < nk) storeL((kt + 1) & 1);        // fill the other buffer
    __syncthreads();
  }

  // C/D layout: VGPR j holds row M = 8*(lane>=16)+j, col N = lane%16
  const int rbase = mBlk + m0 + ((lane >> 4) << 3);
  const int col0  = nBlk + n0 + (lane & 15);
  #pragma unroll
  for (int j = 0; j < 8; ++j) {
    int row = rbase + j;
    if (GUARD_N) {
      if (col0      < N) Y[(size_t)row * N + col0]      = acc0[j];
      if (col0 + 16 < N) Y[(size_t)row * N + col0 + 16] = acc1[j];
    } else {
      Y[(size_t)row * N + col0]      = acc0[j];
      Y[(size_t)row * N + col0 + 16] = acc1[j];
    }
  }
}

// ---------------------------------------------------------------------------
// Token shift: xx = concat(shift, x[:-1]) - x ; out_j = x + xx * mix_j
// ---------------------------------------------------------------------------
__global__ void token_shift_k(const float* __restrict__ x, const float* __restrict__ shift,
                              const float* __restrict__ mr, const float* __restrict__ mw,
                              const float* __restrict__ mk, const float* __restrict__ mv,
                              const float* __restrict__ ma, const float* __restrict__ mg,
                              float* __restrict__ xr, float* __restrict__ xw,
                              float* __restrict__ xk, float* __restrict__ xv,
                              float* __restrict__ xa, float* __restrict__ xg, int total)
{
  int i = blockIdx.x * blockDim.x + threadIdx.x;
  if (i >= total) return;
  int c  = i % C_DIM;
  int bt = i / C_DIM;
  int t  = bt % T_DIM;
  int b  = bt / T_DIM;
  float xi   = x[i];
  float prev = (t == 0) ? shift[b * C_DIM + c] : x[i - C_DIM];
  float xx   = prev - xi;
  xr[i] = xi + xx * mr[c];
  xw[i] = xi + xx * mw[c];
  xk[i] = xi + xx * mk[c];
  xv[i] = xi + xx * mv[c];
  xa[i] = xi + xx * ma[c];
  xg[i] = xi + xx * mg[c];
}

// ---------------------------------------------------------------------------
// Elementwise helpers
// ---------------------------------------------------------------------------
__global__ void tanh_k(float* y, int n) {
  int i = blockIdx.x * blockDim.x + threadIdx.x;
  if (i < n) y[i] = tanhf(y[i]);
}
__global__ void sig_k(float* y, int n) {
  int i = blockIdx.x * blockDim.x + threadIdx.x;
  if (i < n) y[i] = sigmoidf_(y[i]);
}
// w = exp(-softplus(-(w0 + pre)) - 0.5)
__global__ void decay_k(float* y, const float* __restrict__ w0, int n) {
  int i = blockIdx.x * blockDim.x + threadIdx.x;
  if (i >= n) return;
  int c = i % C_DIM;
  float z  = -(w0[c] + y[i]);
  float sp = (z > 20.0f) ? z : log1pf(__expf(z));   // softplus(z)
  y[i] = __expf(-sp - 0.5f);
}
// y = sigmoid(bias[c] + y)
__global__ void bias_sig_k(float* y, const float* __restrict__ b0, int n) {
  int i = blockIdx.x * blockDim.x + threadIdx.x;
  if (i >= n) return;
  int c = i % C_DIM;
  y[i] = sigmoidf_(b0[c] + y[i]);
}
// v = v + (v_first - v) * sigmoid(v0[c] + pre)
__global__ void vmix_k(float* __restrict__ v, const float* __restrict__ vf,
                       const float* __restrict__ pre, const float* __restrict__ v0, int n) {
  int i = blockIdx.x * blockDim.x + threadIdx.x;
  if (i >= n) return;
  int c = i % C_DIM;
  float s  = sigmoidf_(v0[c] + pre[i]);
  float vv = v[i];
  v[i] = vv + (vf[i] - vv) * s;
}

// ---------------------------------------------------------------------------
// Per-head kk-normalize + k2 (one wave32 per (b,t,h), 2 channels/lane)
// ain = -kk_hat ; bin = kk_hat * a ; k <- k*(1+(a-1)*k_a)
// ---------------------------------------------------------------------------
__global__ void kprep_k(float* __restrict__ k, const float* __restrict__ a,
                        const float* __restrict__ kkc, const float* __restrict__ kac,
                        float* __restrict__ ain, float* __restrict__ bin, int totalWaves)
{
  int gw   = (blockIdx.x * blockDim.x + threadIdx.x) >> 5;
  int lane = threadIdx.x & 31;
  if (gw >= totalWaves) return;
  int h = gw % HEADS;
  size_t bt   = (size_t)(gw / HEADS);
  size_t base = bt * C_DIM + (size_t)h * HN;
  int c0 = h * HN + lane, c1 = c0 + 32;

  float k0v = k[base + lane], k1v = k[base + lane + 32];
  float kk0 = k0v * kkc[c0],  kk1 = k1v * kkc[c1];
  float ss  = kk0 * kk0 + kk1 * kk1;
  #pragma unroll
  for (int m = 16; m >= 1; m >>= 1) ss += __shfl_xor(ss, m, 32);
  float inv = 1.0f / fmaxf(sqrtf(ss), 1e-12f);
  float n0v = kk0 * inv, n1v = kk1 * inv;

  float a0v = a[base + lane], a1v = a[base + lane + 32];
  ain[base + lane]      = -n0v;
  ain[base + lane + 32] = -n1v;
  bin[base + lane]      = n0v * a0v;
  bin[base + lane + 32] = n1v * a1v;
  k[base + lane]        = k0v * (1.0f + (a0v - 1.0f) * kac[c0]);
  k[base + lane + 32]   = k1v * (1.0f + (a1v - 1.0f) * kac[c1]);
}

// ---------------------------------------------------------------------------
// Sequential WKV scan. One block per (b,h), 64 threads; thread v keeps state
// row S[v][0..63] in 64 VGPRs. Per step: sa=S.a ; S=S*w+sa*b+v*k ; o=S.r
// ---------------------------------------------------------------------------
__global__ __launch_bounds__(64)
void wkv_scan_k(const float* __restrict__ r, const float* __restrict__ k,
                const float* __restrict__ v, const float* __restrict__ w,
                const float* __restrict__ ain, const float* __restrict__ bin,
                const float* __restrict__ S0, float* __restrict__ o)
{
  const int bh = blockIdx.x;
  const int b  = bh / HEADS, h = bh % HEADS;
  const int vr = threadIdx.x;                  // output row 0..63

  float S[HN];
  const float* s0 = S0 + (((size_t)b * HEADS + h) * HN + vr) * HN;
  #pragma unroll
  for (int j = 0; j < HN; ++j) S[j] = s0[j];

  __shared__ float sr[HN], sk[HN], sv[HN], sw[HN], sa[HN], sb[HN];
  const size_t baseBH = (size_t)b * T_DIM * C_DIM + (size_t)h * HN;

  for (int t = 0; t < T_DIM; ++t) {
    size_t off = baseBH + (size_t)t * C_DIM + vr;
    sr[vr] = r[off];  sk[vr] = k[off];   sv[vr] = v[off];
    sw[vr] = w[off];  sa[vr] = ain[off]; sb[vr] = bin[off];
    __syncthreads();

    float acc = 0.0f;
    #pragma unroll
    for (int j = 0; j < HN; ++j) acc += S[j] * sa[j];
    const float vv = sv[vr];
    float ot = 0.0f;
    #pragma unroll
    for (int j = 0; j < HN; ++j) {
      float sj = S[j] * sw[j] + acc * sb[j] + vv * sk[j];
      S[j] = sj;
      ot += sj * sr[j];
    }
    o[off] = ot;
    __syncthreads();
  }
}

// ---------------------------------------------------------------------------
// GroupNorm(per-head) + rkv bonus + gate. One wave32 per (b,t,h).
// out = (GN(o)*ln_w + ln_b + (sum_n r*k2*r_k)*v) * g
// ---------------------------------------------------------------------------
__global__ void gn_bonus_gate_k(const float* __restrict__ o,  const float* __restrict__ r,
                                const float* __restrict__ k2, const float* __restrict__ v,
                                const float* __restrict__ g,  const float* __restrict__ rk,
                                const float* __restrict__ lnw, const float* __restrict__ lnb,
                                float* __restrict__ out, int totalWaves)
{
  int gw   = (blockIdx.x * blockDim.x + threadIdx.x) >> 5;
  int lane = threadIdx.x & 31;
  if (gw >= totalWaves) return;
  int h = gw % HEADS;
  size_t bt   = (size_t)(gw / HEADS);
  size_t base = bt * C_DIM + (size_t)h * HN;
  int c0 = h * HN + lane, c1 = c0 + 32;

  float o0 = o[base + lane], o1 = o[base + lane + 32];
  float s = o0 + o1;
  #pragma unroll
  for (int m = 16; m >= 1; m >>= 1) s += __shfl_xor(s, m, 32);
  float mu = s * (1.0f / 64.0f);
  float d0 = o0 - mu, d1 = o1 - mu;
  float vs = d0 * d0 + d1 * d1;
  #pragma unroll
  for (int m = 16; m >= 1; m >>= 1) vs += __shfl_xor(vs, m, 32);
  float inv = rsqrtf(vs * (1.0f / 64.0f) + GN_EPS);

  float p = r[base + lane]      * k2[base + lane]      * rk[c0]
          + r[base + lane + 32] * k2[base + lane + 32] * rk[c1];
  #pragma unroll
  for (int m = 16; m >= 1; m >>= 1) p += __shfl_xor(p, m, 32);

  float on0 = d0 * inv * lnw[c0] + lnb[c0];
  float on1 = d1 * inv * lnw[c1] + lnb[c1];
  out[base + lane]      = (on0 + p * v[base + lane])      * g[base + lane];
  out[base + lane + 32] = (on1 + p * v[base + lane + 32]) * g[base + lane + 32];
}

// ---------------------------------------------------------------------------
extern "C" void kernel_launch(void* const* d_in, const int* in_sizes, int n_in,
                              void* d_out, int out_size, void* d_ws, size_t ws_size,
                              hipStream_t stream)
{
  const float* x    = (const float*)d_in[0];
  const float* vf   = (const float*)d_in[1];
  const float* shift= (const float*)d_in[2];
  const float* wkv0 = (const float*)d_in[3];
  const float* mr   = (const float*)d_in[4];
  const float* mw   = (const float*)d_in[5];
  const float* mk   = (const float*)d_in[6];
  const float* mv   = (const float*)d_in[7];
  const float* ma   = (const float*)d_in[8];
  const float* mg   = (const float*)d_in[9];
  const float* w0v  = (const float*)d_in[10];
  const float* w1m  = (const float*)d_in[11];
  const float* w2m  = (const float*)d_in[12];
  const float* a0v  = (const float*)d_in[13];
  const float* a1m  = (const float*)d_in[14];
  const float* a2m  = (const float*)d_in[15];
  const float* v0v  = (const float*)d_in[16];
  const float* v1m  = (const float*)d_in[17];
  const float* v2m  = (const float*)d_in[18];
  const float* g1m  = (const float*)d_in[19];
  const float* g2m  = (const float*)d_in[20];
  const float* kkc  = (const float*)d_in[21];
  const float* kac  = (const float*)d_in[22];
  const float* rkm  = (const float*)d_in[23];
  const float* Wr   = (const float*)d_in[24];
  const float* Wk   = (const float*)d_in[25];
  const float* Wv   = (const float*)d_in[26];
  const float* Wo   = (const float*)d_in[27];
  const float* lnw  = (const float*)d_in[28];
  const float* lnb  = (const float*)d_in[29];

  const size_t SL = (size_t)M_ROWS * C_DIM;    // 4M floats per slab
  float* ws = (float*)d_ws;
  float* xr = ws + 0 * SL;   // later reused: a_in
  float* xw = ws + 1 * SL;   // later: w_pre -> decay w
  float* xk = ws + 2 * SL;   // later: b_in
  float* xv = ws + 3 * SL;   // later: vmix_pre -> pre-Wo activations
  float* xa = ws + 4 * SL;   // later: a (sigmoid)
  float* xg = ws + 5 * SL;   // later: g
  float* rb = ws + 6 * SL;
  float* kb = ws + 7 * SL;   // later: k2 (in place)
  float* vb = ws + 8 * SL;   // later: v (mixed, in place)
  float* ob = ws + 9 * SL;   // scan output
  float* h1 = ws + 10 * SL;              // 2048 x 64
  float* h2 = h1 + (size_t)M_ROWS * 64;  // 2048 x 32
  float* h3 = h2 + (size_t)M_ROWS * 32;  // 2048 x 64
  float* h4 = h3 + (size_t)M_ROWS * 64;  // 2048 x 128

  const int EW_BLOCKS   = (int)((SL + 255) / 256);
  const int WAVES_TOTAL = M_ROWS * HEADS;                  // 65536
  const int WAVE_BLOCKS = (WAVES_TOTAL * 32 + 255) / 256;  // 8192

  auto gemm = [&](const float* X, const float* Wm, float* Y, int M, int N, int K) {
    dim3 grid((N + 63) / 64, M / 64);
    if ((N & 63) == 0)
      gemm_bf16_wmma<false><<<grid, dim3(256), 0, stream>>>(X, Wm, Y, M, N, K);
    else
      gemm_bf16_wmma<true ><<<grid, dim3(256), 0, stream>>>(X, Wm, Y, M, N, K);
  };

  // 1. token shift
  token_shift_k<<<EW_BLOCKS, 256, 0, stream>>>(x, shift, mr, mw, mk, mv, ma, mg,
                                               xr, xw, xk, xv, xa, xg, (int)SL);
  // 2. big projections
  gemm(xr, Wr, rb, M_ROWS, C_DIM, C_DIM);
  gemm(xk, Wk, kb, M_ROWS, C_DIM, C_DIM);
  gemm(xv, Wv, vb, M_ROWS, C_DIM, C_DIM);

  // 3. decay path: w = exp(-softplus(-(w0 + tanh(xw@w1)@w2)) - 0.5)
  gemm(xw, w1m, h1, M_ROWS, 64, C_DIM);
  tanh_k<<<(M_ROWS * 64 + 255) / 256, 256, 0, stream>>>(h1, M_ROWS * 64);
  gemm(h1, w2m, xw, M_ROWS, C_DIM, 64);
  decay_k<<<EW_BLOCKS, 256, 0, stream>>>(xw, w0v, (int)SL);

  // 4. value residual mix
  gemm(xv, v1m, h2, M_ROWS, 32, C_DIM);     // N=32 -> guarded variant
  gemm(h2, v2m, xv, M_ROWS, C_DIM, 32);
  vmix_k<<<EW_BLOCKS, 256, 0, stream>>>(vb, vf, xv, v0v, (int)SL);

  // 5. a = sigmoid(a0 + xa@a1@a2)
  gemm(xa, a1m, h3, M_ROWS, 64, C_DIM);
  gemm(h3, a2m, xa, M_ROWS, C_DIM, 64);
  bias_sig_k<<<EW_BLOCKS, 256, 0, stream>>>(xa, a0v, (int)SL);

  // 6. g = sigmoid(xg@g1)@g2
  gemm(xg, g1m, h4, M_ROWS, 128, C_DIM);
  sig_k<<<(M_ROWS * 128 + 255) / 256, 256, 0, stream>>>(h4, M_ROWS * 128);
  gemm(h4, g2m, xg, M_ROWS, C_DIM, 128);

  // 7. kk-normalize + k2 (a_in -> xr slab, b_in -> xk slab, k2 in-place in kb)
  kprep_k<<<WAVE_BLOCKS, 256, 0, stream>>>(kb, xa, kkc, kac, xr, xk, WAVES_TOTAL);

  // 8. sequential WKV scan
  wkv_scan_k<<<B_DIM * HEADS, 64, 0, stream>>>(rb, kb, vb, xw, xr, xk, wkv0, ob);

  // 9. GroupNorm + bonus + gate -> xv slab
  gn_bonus_gate_k<<<WAVE_BLOCKS, 256, 0, stream>>>(ob, rb, kb, vb, xg, rkm,
                                                   lnw, lnb, xv, WAVES_TOTAL);
  // 10. final projection
  gemm(xv, Wo, (float*)d_out, M_ROWS, C_DIM, C_DIM);
}
